// SMHAdapter_57183194579415
// MI455X (gfx1250) — matrix operand order repaired and checked by reading, not compiled, not run on hardware
//
#include <hip/hip_runtime.h>
#include <hip/hip_bf16.h>

// Problem constants
#define BB   2
#define LL   2048
#define DD   768
#define HIDN 32
#define TOK  (BB * LL)          // 4096 tokens
#define SCALE_F 0.036084391824351615f   // 1/sqrt(768)

// Attention tiling
#define TPB   8                 // tokens per block
#define WROWS 39                // staged window rows: l0-31 .. l0+7
#define RPAD  776               // LDS row stride in bf16 (1552B, 16B aligned, de-banked)

typedef __attribute__((ext_vector_type(16))) __bf16 v16bf;
typedef __attribute__((ext_vector_type(8)))  float  v8f;

__device__ __forceinline__ float sigmoidf_(float x) { return 1.0f / (1.0f + __expf(-x)); }

// 16B global -> LDS async copy (ASYNCcnt-tracked). lds_off is the wave-relative
// LDS byte offset (low 32 bits of a generic pointer to __shared__).
__device__ __forceinline__ void async_copy_b128(unsigned lds_off, const void* gsrc) {
    asm volatile("global_load_async_to_lds_b128 %0, %1, off"
                 :: "v"(lds_off), "v"(gsrc) : "memory");
}
__device__ __forceinline__ void wait_async0() {
    asm volatile("s_wait_asynccnt 0x0" ::: "memory");
}

// ---------------------------------------------------------------------------
// 1) Saliency gate: h = hidden * clip(sigmoid(silu(s*gw1+gb1) @ gw2^T + gb2))
// ---------------------------------------------------------------------------
__global__ void __launch_bounds__(256)
gate_kernel(const float* __restrict__ hs, const float* __restrict__ sal,
            const float* __restrict__ gw1, const float* __restrict__ gb1,
            const float* __restrict__ gw2, const float* __restrict__ gb2,
            float* __restrict__ h, __bf16* __restrict__ hb) {
    const int l = blockIdx.x, b = blockIdx.y;
    const int t = b * LL + l;
    __shared__ float h1s[HIDN];
    const int tid = threadIdx.x;
    const float s = sal[t];
    if (tid < HIDN) {
        float x = s * gw1[tid] + gb1[tid];   // gw1 is (HID,1)
        h1s[tid] = x * sigmoidf_(x);         // silu
    }
    __syncthreads();
#pragma unroll
    for (int i = 0; i < 3; ++i) {
        const int d = tid + i * 256;
        float acc = gb2[d];
#pragma unroll 8
        for (int j = 0; j < HIDN; ++j) acc += h1s[j] * gw2[d * HIDN + j];
        float g = sigmoidf_(acc);
        g = fminf(fmaxf(g, 0.05f), 0.95f);
        const float hv = hs[(size_t)t * DD + d] * g;
        h[(size_t)t * DD + d]  = hv;
        hb[(size_t)t * DD + d] = (__bf16)hv;
    }
}

// ---------------------------------------------------------------------------
// 2) fp32 -> bf16 conversion
// ---------------------------------------------------------------------------
__global__ void to_bf16_kernel(const float* __restrict__ src, __bf16* __restrict__ dst, int n) {
    int i = blockIdx.x * blockDim.x + threadIdx.x;
    if (i < n) dst[i] = (__bf16)src[i];
}

// ---------------------------------------------------------------------------
// 3) GEMM: out[m,n] = sum_k A[m,k]*W[n,k] + bias[n] (+ resid), bf16 WMMA,
//    fp32 accum. Optional fp32 and/or bf16 outputs.
// ---------------------------------------------------------------------------
__global__ void __launch_bounds__(256)
gemm_bf16_kernel(const __bf16* __restrict__ A, const __bf16* __restrict__ W,
                 const float* __restrict__ bias, const float* __restrict__ resid,
                 float* __restrict__ outf, __bf16* __restrict__ outb) {
    const int K = DD;
    const int n0 = blockIdx.x * 64;
    const int m0 = blockIdx.y * 64;
    const int tid   = threadIdx.x;
    const int wave  = tid >> 5;
    const int lane  = tid & 31;
    const int lan15 = lane & 15;
    const int khalf = lane >> 4;          // 0 or 1
    const int mi    = wave & 3;           // M sub-tile 0..3
    const int njb   = (wave >> 2) * 2;    // N sub-tile base {0,2}

    const int arow  = m0 + mi * 16 + lan15;
    const int bcol0 = n0 + njb * 16 + lan15;
    const int bcol1 = bcol0 + 16;

    const __bf16* aBase  = A + (size_t)arow  * K;
    const __bf16* b0Base = W + (size_t)bcol0 * K;
    const __bf16* b1Base = W + (size_t)bcol1 * K;

    v8f c0 = {};
    v8f c1 = {};

    for (int k0 = 0; k0 < K; k0 += 32) {
        if (k0 + 64 <= K) {   // WGP-scope temporal prefetch of next K tile
            __builtin_prefetch(aBase  + k0 + 32, 0, 3);
            __builtin_prefetch(b0Base + k0 + 32, 0, 3);
            __builtin_prefetch(b1Base + k0 + 32, 0, 3);
        }
        // A 16x32 bf16 frag: lanes 0-15 -> K 0..15; lanes 16-31 -> K 16..31
        v16bf a, b0, b1;
        const __bf16* ap0 = aBase + k0 + khalf * 16;
        const __bf16* ap1 = ap0 + 8;
#pragma unroll
        for (int e = 0; e < 8; ++e) { a[e] = ap0[e]; a[8 + e] = ap1[e]; }
        // B 32x16 bf16 frag: lanes 0-15 -> K 0..15; lanes 16-31 -> K 16..31
        const __bf16* bp0 = b0Base + k0 + khalf * 16;
        const __bf16* bp1 = b1Base + k0 + khalf * 16;
#pragma unroll
        for (int e = 0; e < 16; ++e) { b0[e] = bp0[e]; b1[e] = bp1[e]; }

        c0 = __builtin_amdgcn_wmma_f32_16x16x32_bf16(false, a, false, b0, (short)0, c0, false, false);
        c1 = __builtin_amdgcn_wmma_f32_16x16x32_bf16(false, a, false, b1, (short)0, c1, false, false);
    }

    // C/D layout: vgpr r -> M = r + 8*khalf; N = lane&15
    const int   mrow = m0 + mi * 16 + khalf * 8;
    const float bia0 = bias[bcol0];
    const float bia1 = bias[bcol1];
#pragma unroll
    for (int r = 0; r < 8; ++r) {
        const int m = mrow + r;
        const size_t i0 = (size_t)m * DD + bcol0;
        const size_t i1 = (size_t)m * DD + bcol1;
        float o0 = c0[r] + bia0;
        float o1 = c1[r] + bia1;
        if (resid) { o0 += resid[i0]; o1 += resid[i1]; }
        if (outf)  { outf[i0] = o0;          outf[i1] = o1; }
        if (outb)  { outb[i0] = (__bf16)o0;  outb[i1] = (__bf16)o1; }
    }
}

// ---------------------------------------------------------------------------
// 4) Dual-window causal local attention, 8 tokens/block. The 39-row k/v
//    window (bf16) + 8 q rows (fp32) are staged into LDS with
//    global_load_async_to_lds_b128. Wave w owns token w; lane j computes
//    score j; both softmaxes run in-wave via shfl_xor (w=8 window is the
//    aligned lane octet 24..31). Out-of-range rows use bk/bv (bf16 copies),
//    matching the reference's bias padding.
// ---------------------------------------------------------------------------
__global__ void __launch_bounds__(256)
attn_kernel(const float* __restrict__ q, const __bf16* __restrict__ kb,
            const __bf16* __restrict__ vb, const __bf16* __restrict__ bkb,
            const __bf16* __restrict__ bvb, __bf16* __restrict__ ctxb) {
    __shared__ __align__(16) __bf16 ks[WROWS * RPAD];   // 60528 B
    __shared__ __align__(16) __bf16 vs[WROWS * RPAD];   // 60528 B
    __shared__ __align__(16) float  qs[TPB * DD];       // 24576 B
    __shared__ float wsh[TPB * 32];                     //  1024 B

    const int l0  = blockIdx.x * TPB;
    const int b   = blockIdx.y;
    const int t0  = b * LL + l0;
    const int tid = threadIdx.x;

    // ---- async stage: q (1536 chunks) + k (3744) + v (3744), 16B each ----
    const unsigned qs_base = (unsigned)(size_t)(void*)qs;
    const unsigned ks_base = (unsigned)(size_t)(void*)ks;
    const unsigned vs_base = (unsigned)(size_t)(void*)vs;
    const int QCH = TPB * (DD * 4 / 16);          // 1536
    const int KCH = WROWS * (DD * 2 / 16);        // 3744
    for (int c = tid; c < QCH + 2 * KCH; c += 256) {
        if (c < QCH) {
            const int row = c / 192, ch = c % 192;        // 192 chunks per fp32 row
            async_copy_b128(qs_base + (unsigned)(row * DD * 4 + ch * 16),
                            q + (size_t)(t0 + row) * DD + ch * 4);
        } else {
            const int cc = c - QCH;
            const int tensor = cc >= KCH;                 // 0=k, 1=v
            const int ci  = tensor ? cc - KCH : cc;
            const int row = ci / 96, ch = ci % 96;        // 96 chunks per bf16 row
            const int idx = l0 - 31 + row;
            const __bf16* src;
            if (tensor == 0) src = (idx >= 0) ? kb + (size_t)(b * LL + idx) * DD : bkb;
            else             src = (idx >= 0) ? vb + (size_t)(b * LL + idx) * DD : bvb;
            const unsigned base = tensor ? vs_base : ks_base;
            async_copy_b128(base + (unsigned)(row * RPAD * 2 + ch * 16), src + ch * 8);
        }
    }
    wait_async0();
    __syncthreads();

    // ---- scores: wave = token, lane = window position ----
    const int tok  = tid >> 5;      // 0..7
    const int lane = tid & 31;      // window position j
    const int r    = tok + lane;    // staged row index, 0..38
    const float* qrow = &qs[tok * DD];
    const __bf16* krow = &ks[r * RPAD];
    float s = 0.f;
#pragma unroll 8
    for (int d = 0; d < DD; ++d) s += qrow[d] * (float)krow[d];
    s *= SCALE_F;

    // softmax over 32 (w=32)
    float m = s;
#pragma unroll
    for (int off = 16; off >= 1; off >>= 1) m = fmaxf(m, __shfl_xor(m, off, 32));
    const float e = __expf(s - m);
    float sum = e;
#pragma unroll
    for (int off = 16; off >= 1; off >>= 1) sum += __shfl_xor(sum, off, 32);
    const float a32 = e / sum;
    // softmax over aligned octet (w=8, lanes 24..31 are the live window)
    float m8 = s;
    m8 = fmaxf(m8, __shfl_xor(m8, 1, 32));
    m8 = fmaxf(m8, __shfl_xor(m8, 2, 32));
    m8 = fmaxf(m8, __shfl_xor(m8, 4, 32));
    const float e8 = __expf(s - m8);
    float s8 = e8;
    s8 += __shfl_xor(s8, 1, 32);
    s8 += __shfl_xor(s8, 2, 32);
    s8 += __shfl_xor(s8, 4, 32);
    const float a8 = (lane >= 24) ? (e8 / s8) : 0.f;

    wsh[tok * 32 + lane] = 0.5f * (a32 + a8);
    __syncthreads();

    // ---- context: thread (tok, lane) accumulates 24 dims ----
    float acc[24];
#pragma unroll
    for (int i = 0; i < 24; ++i) acc[i] = 0.f;
    for (int j = 0; j < 32; ++j) {
        const float a = wsh[tok * 32 + j];
        const __bf16* vr = &vs[(tok + j) * RPAD];
#pragma unroll
        for (int i = 0; i < 24; ++i) acc[i] += a * (float)vr[lane + i * 32];
    }
    const size_t base = (size_t)(t0 + tok) * DD;
#pragma unroll
    for (int i = 0; i < 24; ++i) ctxb[base + lane + i * 32] = (__bf16)acc[i];
}

// ---------------------------------------------------------------------------
// Workspace layout (bytes)
// ---------------------------------------------------------------------------
static const size_t OFF_H    = 0;                        // h fp32      12.58 MB
static const size_t OFF_HB   = 12582912;                 // h bf16       6.29 MB
static const size_t OFF_Q    = 18874368;                 // q fp32      12.58 MB
static const size_t OFF_KB   = 31457280;                 // k bf16       6.29 MB
static const size_t OFF_VB   = 37748736;                 // v bf16       6.29 MB
static const size_t OFF_CTXB = 44040192;                 // ctx bf16     6.29 MB
static const size_t OFF_WQB  = 50331648;                 // 4x 768^2 bf16
static const size_t OFF_WKB  = OFF_WQB + 1179648;
static const size_t OFF_WVB  = OFF_WKB + 1179648;
static const size_t OFF_WOB  = OFF_WVB + 1179648;
static const size_t OFF_BKB  = OFF_WOB + 1179648;        // bk bf16 (2KB slot)
static const size_t OFF_BVB  = OFF_BKB + 2048;           // bv bf16; end ~55.05 MB

extern "C" void kernel_launch(void* const* d_in, const int* in_sizes, int n_in,
                              void* d_out, int out_size, void* d_ws, size_t ws_size,
                              hipStream_t stream) {
    (void)in_sizes; (void)n_in; (void)out_size; (void)ws_size;
    const float* hs  = (const float*)d_in[0];
    const float* sal = (const float*)d_in[1];
    const float* gw1 = (const float*)d_in[2];
    const float* gb1 = (const float*)d_in[3];
    const float* gw2 = (const float*)d_in[4];
    const float* gb2 = (const float*)d_in[5];
    const float* wq  = (const float*)d_in[6];
    const float* bq  = (const float*)d_in[7];
    const float* wk  = (const float*)d_in[8];
    const float* bk  = (const float*)d_in[9];
    const float* wv  = (const float*)d_in[10];
    const float* bv  = (const float*)d_in[11];
    const float* wo  = (const float*)d_in[12];
    const float* bo  = (const float*)d_in[13];

    char* ws = (char*)d_ws;
    float*  h    = (float*)(ws + OFF_H);
    __bf16* hb   = (__bf16*)(ws + OFF_HB);
    float*  qf   = (float*)(ws + OFF_Q);
    __bf16* kbb  = (__bf16*)(ws + OFF_KB);
    __bf16* vbb  = (__bf16*)(ws + OFF_VB);
    __bf16* ctxb = (__bf16*)(ws + OFF_CTXB);
    __bf16* wqb  = (__bf16*)(ws + OFF_WQB);
    __bf16* wkb  = (__bf16*)(ws + OFF_WKB);
    __bf16* wvb  = (__bf16*)(ws + OFF_WVB);
    __bf16* wob  = (__bf16*)(ws + OFF_WOB);
    __bf16* bkb  = (__bf16*)(ws + OFF_BKB);
    __bf16* bvb  = (__bf16*)(ws + OFF_BVB);
    float*  out  = (float*)d_out;

    // 1) gate + h (fp32 + bf16)
    gate_kernel<<<dim3(LL, BB), 256, 0, stream>>>(hs, sal, gw1, gb1, gw2, gb2, h, hb);

    // 2) weight + bias conversions
    const int WN = DD * DD;
    const int cb = (WN + 255) / 256;
    to_bf16_kernel<<<cb, 256, 0, stream>>>(wq, wqb, WN);
    to_bf16_kernel<<<cb, 256, 0, stream>>>(wk, wkb, WN);
    to_bf16_kernel<<<cb, 256, 0, stream>>>(wv, wvb, WN);
    to_bf16_kernel<<<cb, 256, 0, stream>>>(wo, wob, WN);
    to_bf16_kernel<<<3, 256, 0, stream>>>(bk, bkb, DD);
    to_bf16_kernel<<<3, 256, 0, stream>>>(bv, bvb, DD);

    // 3) Q/K/V projections via bf16 WMMA (q -> fp32; k,v -> bf16 only)
    const dim3 ggrid(DD / 64, TOK / 64);   // (12, 64)
    gemm_bf16_kernel<<<ggrid, 256, 0, stream>>>(hb, wqb, bq, nullptr, qf, nullptr);
    gemm_bf16_kernel<<<ggrid, 256, 0, stream>>>(hb, wkb, bk, nullptr, nullptr, kbb);
    gemm_bf16_kernel<<<ggrid, 256, 0, stream>>>(hb, wvb, bv, nullptr, nullptr, vbb);

    // 4) dual-window local attention (LDS-staged via async copies)
    attn_kernel<<<dim3(LL / TPB, BB), 256, 0, stream>>>(qf, kbb, vbb, bkb, bvb, ctxb);

    // 5) out = h + ctx @ wo^T + bo
    gemm_bf16_kernel<<<ggrid, 256, 0, stream>>>(ctxb, wob, bo, h, out, nullptr);
}
